// CausalSelfAttention_23819888624336
// MI455X (gfx1250) — compile-verified, matrix operand order checked by reading, not run on hardware
//
#include <hip/hip_runtime.h>
#include <hip/hip_bf16.h>

typedef unsigned short u16;
typedef unsigned int   u32;
typedef __attribute__((ext_vector_type(16))) __bf16       v16bf;
typedef __attribute__((ext_vector_type(8)))  float        v8f;
typedef __attribute__((ext_vector_type(8)))  unsigned int v8u;

#define N_EMBD   1024
#define N_HEAD   16
#define HEAD_DIM 64
#define SEQ      2048
#define BATCH    2
#define ROWS     (BATCH * SEQ)     /* 4096 */
#define QKV_COLS (3 * N_EMBD)      /* 3072 */

// ---------------- helpers ----------------

__device__ __forceinline__ u16 f2bf(float f) {
  u32 u = __builtin_bit_cast(u32, f);
  u += 0x7FFFu + ((u >> 16) & 1u);          // round-to-nearest-even
  return (u16)(u >> 16);
}

__device__ __forceinline__ v8f vzero8() {
  v8f z;
#pragma unroll
  for (int i = 0; i < 8; ++i) z[i] = 0.0f;
  return z;
}

__device__ __forceinline__ v8f wmma_bf16(v16bf a, v16bf b, v8f c) {
  // D = A(16x32 bf16) x B(32x16 bf16) + C(16x16 f32)
  return __builtin_amdgcn_wmma_f32_16x16x32_bf16(
      /*neg_a=*/false, a, /*neg_b=*/false, b,
      /*c_mod=*/(short)0, c, /*reuse_a=*/false, /*reuse_b=*/false);
}

// Load one 16x32-bf16 fragment (A or B operand) for this lane.
// 16-bit layout: vgpr i holds K = (i>=4?16:0) + half*8 + (i&3)*2 .. +1
// dword indices form two contiguous runs of 4 -> coalesces to 2x b128.
__device__ __forceinline__ v16bf frag_ld(const u32* base, int khalf) {
  v8u u;
#pragma unroll
  for (int i = 0; i < 8; ++i)
    u[i] = base[(i & 3) + (khalf ? 4 : 0) + ((i & 4) ? 8 : 0)];
  return __builtin_bit_cast(v16bf, u);
}

// xor-swizzle within 32 lanes (and_mask=0x1F, or=0, xor=MASK)
template <int MASK>
__device__ __forceinline__ float swz_xor(float v) {
  int r = __builtin_amdgcn_ds_swizzle(__builtin_bit_cast(int, v),
                                      (MASK << 10) | 0x1F);
  return __builtin_bit_cast(float, r);
}
__device__ __forceinline__ float red16_max(float v) {
  v = fmaxf(v, swz_xor<1>(v)); v = fmaxf(v, swz_xor<2>(v));
  v = fmaxf(v, swz_xor<4>(v)); v = fmaxf(v, swz_xor<8>(v));
  return v;
}
__device__ __forceinline__ float red16_sum(float v) {
  v += swz_xor<1>(v); v += swz_xor<2>(v);
  v += swz_xor<4>(v); v += swz_xor<8>(v);
  return v;
}

// one pipelined GEMM fragment set: A row tile + 4 B column tiles
struct Frags { v16bf a, x0, x1, x2, x3; };

// ---------------- conversion kernels ----------------

__global__ void k_cvt(const float* __restrict__ src, u16* __restrict__ dst, int n) {
  int i = blockIdx.x * 256 + threadIdx.x;
  if (i < n) dst[i] = f2bf(src[i]);
}

// src [rows][cols] fp32  ->  dst [cols][rows] bf16  (coalesced reads)
__global__ void k_cvt_t(const float* __restrict__ src, u16* __restrict__ dst,
                        int rows, int cols) {
  long i = (long)blockIdx.x * 256 + threadIdx.x;
  long n = (long)rows * cols;
  if (i < n) {
    int c = (int)(i % cols);
    int r = (int)(i / cols);
    dst[(long)c * rows + r] = f2bf(src[i]);
  }
}

// ping-pong pipelined K=1024 GEMM core (32 k-steps of 32 bf16)
#define LOADSET(F, OFF)                                  \
  do {                                                   \
    (F).a  = frag_ld(ab + (OFF), khalf);                 \
    (F).x0 = frag_ld(b0 + (OFF), khalf);                 \
    (F).x1 = frag_ld(b1 + (OFF), khalf);                 \
    (F).x2 = frag_ld(b2 + (OFF), khalf);                 \
    (F).x3 = frag_ld(b3 + (OFF), khalf);                 \
  } while (0)

#define MMASET(F)                                        \
  do {                                                   \
    c0 = wmma_bf16((F).a, (F).x0, c0);                   \
    c1 = wmma_bf16((F).a, (F).x1, c1);                   \
    c2 = wmma_bf16((F).a, (F).x2, c2);                   \
    c3 = wmma_bf16((F).a, (F).x3, c3);                   \
  } while (0)

#define GEMM_CORE()                                      \
  Frags fA, fB;                                          \
  LOADSET(fA, 0);                                        \
  for (int off = 0; off < 480 - 32; off += 32) {         \
    __builtin_prefetch(ab + off + 64, 0, 1);             \
    LOADSET(fB, off + 16);                               \
    MMASET(fA);                                          \
    LOADSET(fA, off + 32);                               \
    MMASET(fB);                                          \
  }                                                      \
  /* loop ends at off == 448; fA holds step 448; handle tail */\
  LOADSET(fB, 464);                                      \
  MMASET(fA);          /* step 448 */                    \
  LOADSET(fA, 480);                                      \
  MMASET(fB);          /* step 464 */                    \
  LOADSET(fB, 496);                                      \
  MMASET(fA);          /* step 480 */                    \
  MMASET(fB);          /* step 496 */

// ---------------- GEMM 1: qkv = x @ Wqkv + b, scatter to heads ----------------

__global__ __launch_bounds__(128)
void k_gemm_qkv(const u16* __restrict__ Xb, const u16* __restrict__ Wt,
                const float* __restrict__ bias,
                u16* __restrict__ Qb, u16* __restrict__ Kb, u16* __restrict__ Vt) {
  const int wave = threadIdx.x >> 5, lane = threadIdx.x & 31;
  const int lane16 = lane & 15, khalf = lane >> 4;
  const int tile = blockIdx.x * 4 + wave;
  const int colt = tile % (QKV_COLS / 64);
  const int rowt = tile / (QKV_COLS / 64);
  const int r0 = rowt * 16, n0 = colt * 64;

  v8f c0 = vzero8(), c1 = vzero8(), c2 = vzero8(), c3 = vzero8();
  const u32* ab = (const u32*)(Xb + (size_t)(r0 + lane16) * N_EMBD);
  const u32* b0 = (const u32*)(Wt + (size_t)(n0 +  0 + lane16) * N_EMBD);
  const u32* b1 = (const u32*)(Wt + (size_t)(n0 + 16 + lane16) * N_EMBD);
  const u32* b2 = (const u32*)(Wt + (size_t)(n0 + 32 + lane16) * N_EMBD);
  const u32* b3 = (const u32*)(Wt + (size_t)(n0 + 48 + lane16) * N_EMBD);

  GEMM_CORE();

  const int which = n0 >> 10;   // 0:Q 1:K 2:V (uniform per wave)
  v8f cc[4] = {c0, c1, c2, c3};
#pragma unroll
  for (int nt = 0; nt < 4; ++nt) {
    const int ocol = n0 + nt * 16 + lane16;
    const float bv = bias[ocol];
    const int col1024 = ocol & (N_EMBD - 1);
    const int h = col1024 >> 6, d = col1024 & 63;
#pragma unroll
    for (int r = 0; r < 8; ++r) {
      const int tg = r0 + r + (khalf ? 8 : 0);       // global row 0..4095
      const int bidx = tg >> 11, tin = tg & (SEQ - 1);
      const u16 val = f2bf(cc[nt][r] + bv);
      if (which == 0)
        Qb[((size_t)(bidx * N_HEAD + h) * SEQ + tin) * HEAD_DIM + d] = val;
      else if (which == 1)
        Kb[((size_t)(bidx * N_HEAD + h) * SEQ + tin) * HEAD_DIM + d] = val;
      else
        Vt[((size_t)(bidx * N_HEAD + h) * HEAD_DIM + d) * SEQ + tin] = val;
    }
  }
}

// ---------------- flash attention (no mask, scale = 1/sqrt(H) = 0.25) ----------------

struct KF { v16bf f0a, f0b, f1a, f1b; };

__device__ __forceinline__ void load_kf(KF& k, const u16* Kh, int kv,
                                        int lane16, int khalf) {
  const u32* kp0 = (const u32*)(Kh + (size_t)(kv + lane16) * HEAD_DIM);
  const u32* kp1 = (const u32*)(Kh + (size_t)(kv + 16 + lane16) * HEAD_DIM);
  k.f0a = frag_ld(kp0, khalf); k.f0b = frag_ld(kp0 + 16, khalf);
  k.f1a = frag_ld(kp1, khalf); k.f1b = frag_ld(kp1 + 16, khalf);
}

__global__ __launch_bounds__(128)
void k_attn(const u16* __restrict__ Qb, const u16* __restrict__ Kb,
            const u16* __restrict__ Vt, u16* __restrict__ Ob) {
  __shared__ u16 Plds[4][16][40];                  // per-wave 16x32 P tile (+pad)
  const int wave = threadIdx.x >> 5, lane = threadIdx.x & 31;
  const int lane16 = lane & 15, khalf = lane >> 4;
  const int id = blockIdx.x * 4 + wave;
  const int qt = id & 127;                          // 128 query tiles
  const int bh = id >> 7;                           // 0..31 (b*H+h)
  const int q0 = qt * 16;
  const u16* Qh = Qb + (size_t)bh * SEQ * HEAD_DIM;
  const u16* Kh = Kb + (size_t)bh * SEQ * HEAD_DIM;
  const u16* Vh = Vt + (size_t)bh * HEAD_DIM * SEQ;

  const u32* qbase = (const u32*)(Qh + (size_t)(q0 + lane16) * HEAD_DIM);
  const v16bf aq0 = frag_ld(qbase, khalf);          // k = 0..31
  const v16bf aq1 = frag_ld(qbase + 16, khalf);     // k = 32..63

  float m[8], lsum[8];
  v8f acc[4];
#pragma unroll
  for (int r = 0; r < 8; ++r) { m[r] = -3.0e38f; lsum[r] = 0.0f; }
#pragma unroll
  for (int nt = 0; nt < 4; ++nt) acc[nt] = vzero8();

  KF ka, kbuf;
  load_kf(ka, Kh, 0, lane16, khalf);

  // one 32-key block: compute with `cur` K frags, prefetch next into `nxt`
  auto block = [&](KF& cur, KF& nxt, int kv) {
    // ---- S = Q K^T for 32 keys (fragments already resident) ----
    v8f s0 = vzero8(), s1 = vzero8();
    s0 = wmma_bf16(aq0, cur.f0a, s0);
    s0 = wmma_bf16(aq1, cur.f0b, s0);
    s1 = wmma_bf16(aq0, cur.f1a, s1);
    s1 = wmma_bf16(aq1, cur.f1b, s1);

    // ---- issue V fragment loads; exp/swizzle work hides their latency ----
    v16bf vf[4];
#pragma unroll
    for (int nt = 0; nt < 4; ++nt) {
      const u32* vb = (const u32*)(Vh + (size_t)(nt * 16 + lane16) * SEQ + kv);
      vf[nt] = frag_ld(vb, khalf);
    }

    // ---- online softmax (row stats via 16-lane xor reductions) ----
    float p0[8], p1[8], alpha[8];
#pragma unroll
    for (int r = 0; r < 8; ++r) {
      const float a0 = s0[r] * 0.25f, a1 = s1[r] * 0.25f;
      const float rm = red16_max(fmaxf(a0, a1));
      const float mn = fmaxf(m[r], rm);
      alpha[r] = __expf(m[r] - mn);
      m[r] = mn;
      p0[r] = __expf(a0 - mn);
      p1[r] = __expf(a1 - mn);
      const float rs = red16_sum(p0[r] + p1[r]);
      lsum[r] = lsum[r] * alpha[r] + rs;
    }
#pragma unroll
    for (int nt = 0; nt < 4; ++nt)
#pragma unroll
      for (int r = 0; r < 8; ++r) acc[nt][r] *= alpha[r];

    // ---- transpose P through per-wave LDS into A-fragment layout ----
#pragma unroll
    for (int r = 0; r < 8; ++r) {
      const int row = r + (khalf ? 8 : 0);
      Plds[wave][row][lane16]      = f2bf(p0[r]);
      Plds[wave][row][16 + lane16] = f2bf(p1[r]);
    }
    asm volatile("" ::: "memory");    // same-wave DS ops are in-order in HW
    const u32* pb = (const u32*)&Plds[wave][lane16][0];
    const v16bf ap = frag_ld(pb, khalf);

    // ---- prefetch next key block's K fragments (wrapped, branch-free) ----
    load_kf(nxt, Kh, (kv + 32) & (SEQ - 1), lane16, khalf);

    // ---- O += P @ V  (V stored transposed: row = d, contiguous in t) ----
#pragma unroll
    for (int nt = 0; nt < 4; ++nt)
      acc[nt] = wmma_bf16(ap, vf[nt], acc[nt]);
  };

  for (int kv = 0; kv < SEQ; kv += 64) {   // 64 blocks total: even -> ping-pong
    block(ka, kbuf, kv);
    block(kbuf, ka, kv + 32);
  }

  // ---- normalize and store in [B*T, C] bf16 layout for the output GEMM ----
  const int bidx = bh >> 4, h = bh & 15;
#pragma unroll
  for (int nt = 0; nt < 4; ++nt)
#pragma unroll
    for (int r = 0; r < 8; ++r) {
      const int trow = q0 + r + (khalf ? 8 : 0);
      const size_t grow = (size_t)bidx * SEQ + trow;
      const int col = h * HEAD_DIM + nt * 16 + lane16;
      Ob[grow * N_EMBD + col] = f2bf(acc[nt][r] / lsum[r]);
    }
}

// ---------------- GEMM 2: out = attn @ Wout + b (fp32 result) ----------------

__global__ __launch_bounds__(128)
void k_gemm_out(const u16* __restrict__ Ab, const u16* __restrict__ Wt,
                const float* __restrict__ bias, float* __restrict__ out) {
  const int wave = threadIdx.x >> 5, lane = threadIdx.x & 31;
  const int lane16 = lane & 15, khalf = lane >> 4;
  const int tile = blockIdx.x * 4 + wave;
  const int colt = tile % (N_EMBD / 64);
  const int rowt = tile / (N_EMBD / 64);
  const int r0 = rowt * 16, n0 = colt * 64;

  v8f c0 = vzero8(), c1 = vzero8(), c2 = vzero8(), c3 = vzero8();
  const u32* ab = (const u32*)(Ab + (size_t)(r0 + lane16) * N_EMBD);
  const u32* b0 = (const u32*)(Wt + (size_t)(n0 +  0 + lane16) * N_EMBD);
  const u32* b1 = (const u32*)(Wt + (size_t)(n0 + 16 + lane16) * N_EMBD);
  const u32* b2 = (const u32*)(Wt + (size_t)(n0 + 32 + lane16) * N_EMBD);
  const u32* b3 = (const u32*)(Wt + (size_t)(n0 + 48 + lane16) * N_EMBD);

  GEMM_CORE();

  v8f cc[4] = {c0, c1, c2, c3};
#pragma unroll
  for (int nt = 0; nt < 4; ++nt) {
    const int col = n0 + nt * 16 + lane16;
    const float bv = bias[col];
#pragma unroll
    for (int r = 0; r < 8; ++r) {
      const int row = r0 + r + (khalf ? 8 : 0);
      out[(size_t)row * N_EMBD + col] = cc[nt][r] + bv;
    }
  }
}

// ---------------- host launcher ----------------

extern "C" void kernel_launch(void* const* d_in, const int* in_sizes, int n_in,
                              void* d_out, int out_size, void* d_ws, size_t ws_size,
                              hipStream_t stream) {
  const float* x     = (const float*)d_in[0];
  const float* W_qkv = (const float*)d_in[1];
  const float* b_qkv = (const float*)d_in[2];
  const float* W_out = (const float*)d_in[3];
  const float* b_out = (const float*)d_in[4];
  float* out = (float*)d_out;

  char* ws = (char*)d_ws;
  u16* Xb    = (u16*)ws; ws += (size_t)ROWS * N_EMBD * 2;                       // 8 MiB
  u16* Wqkvt = (u16*)ws; ws += (size_t)QKV_COLS * N_EMBD * 2;                   // 6 MiB
  u16* Woutt = (u16*)ws; ws += (size_t)N_EMBD * N_EMBD * 2;                     // 2 MiB
  u16* Qb    = (u16*)ws; ws += (size_t)BATCH * N_HEAD * SEQ * HEAD_DIM * 2;     // 8 MiB
  u16* Kb    = (u16*)ws; ws += (size_t)BATCH * N_HEAD * SEQ * HEAD_DIM * 2;     // 8 MiB
  u16* Vt    = (u16*)ws; ws += (size_t)BATCH * N_HEAD * HEAD_DIM * SEQ * 2;     // 8 MiB
  u16* Ob    = (u16*)ws;                                                        // 8 MiB

  k_cvt  <<<(ROWS * N_EMBD) / 256, 256, 0, stream>>>(x, Xb, ROWS * N_EMBD);
  k_cvt_t<<<(N_EMBD * QKV_COLS) / 256, 256, 0, stream>>>(W_qkv, Wqkvt, N_EMBD, QKV_COLS);
  k_cvt_t<<<(N_EMBD * N_EMBD) / 256, 256, 0, stream>>>(W_out, Woutt, N_EMBD, N_EMBD);

  k_gemm_qkv<<<(ROWS / 16) * (QKV_COLS / 64) / 4, 128, 0, stream>>>(
      Xb, Wqkvt, b_qkv, Qb, Kb, Vt);

  k_attn<<<(BATCH * N_HEAD) * (SEQ / 16) / 4, 128, 0, stream>>>(Qb, Kb, Vt, Ob);

  k_gemm_out<<<(ROWS / 16) * (N_EMBD / 64) / 4, 128, 0, stream>>>(
      Ob, Woutt, b_out, out);
}
